// LlamaAttention_80195629351177
// MI455X (gfx1250) — matrix operand order repaired and checked by reading, not compiled, not run on hardware
//
#include <hip/hip_runtime.h>
#include <hip/hip_bf16.h>

typedef __attribute__((ext_vector_type(16))) __bf16 v16bf;
typedef __attribute__((ext_vector_type(8)))  float  v8f;

// Problem constants (from reference)
constexpr int Bb   = 32;            // batch
constexpr int Hh   = 4096;          // hidden
constexpr int NHq  = 32;            // query heads
constexpr int NKVh = 8;             // kv heads
constexpr int Dd   = 128;           // head dim
constexpr int GRP  = 4;             // NH / NKV
constexpr int BSz  = 16;            // page (block) size
constexpr int MBt  = 128;           // max blocks per seq
constexpr int QKV_COLS = (NHq + 2 * NKVh) * Dd;   // 6144

// ---------------------------------------------------------------------------
// fp32-in / fp32-out GEMM for M == 32 on the WMMA pipe (bf16 mul, f32 acc).
// One wave owns one 16-wide N strip and BOTH 16-row M tiles: the B (weight)
// operand is loaded exactly once per element and feeds two back-to-back
// WMMAs, halving the dominant HBM traffic vs one-tile-per-wave.
// Per-lane element addressing follows the CDNA5 ISA 16-bit A(16x32) /
// B(32x16) / f32 C(16x16) VGPR layouts.
// ---------------------------------------------------------------------------
__global__ __launch_bounds__(128) void wmma_gemm_bf16_m32(
    const float* __restrict__ A, const float* __restrict__ Bm,
    float* __restrict__ C, int N, int K, int lda, int ldb, int ldc)
{
    const int waveId = threadIdx.x >> 5;
    const int lane   = threadIdx.x & 31;
    const int tn     = blockIdx.x * 4 + waveId;
    if (tn * 16 >= N) return;

    const int lo = lane & 15;
    const int hi = lane >> 4;

    const int akoff = hi ? 8 : 0;          // A: lanes 16-31 carry K+8
    const int ncol  = tn * 16 + lo;        // B: lane holds column N=lo
    const int bkoff = hi ? 8 : 0;          // B: lanes 16-31 carry K+8

    const float* Arow0 = A + (size_t)lo * lda;          // M tile 0: rows 0..15
    const float* Arow1 = A + (size_t)(16 + lo) * lda;   // M tile 1: rows 16..31

    v8f acc0 = {}, acc1 = {};
    for (int k0 = 0; k0 < K; k0 += 32) {
        if (k0 + 32 < K)   // prefetch next weight tile (global_prefetch_b8)
            __builtin_prefetch(&Bm[(size_t)(k0 + 32) * ldb + ncol], 0, 1);

        v16bf a0, a1, bv;
#pragma unroll
        for (int r = 0; r < 8; ++r) {
            // A VGPR r: K = 2r (r<4) or 8+2r (r>=4), plus half offset
            const int ka = k0 + ((r < 4) ? (2 * r) : (8 + 2 * r)) + akoff;
            a0[2 * r]     = (__bf16)Arow0[ka];
            a0[2 * r + 1] = (__bf16)Arow0[ka + 1];
            a1[2 * r]     = (__bf16)Arow1[ka];
            a1[2 * r + 1] = (__bf16)Arow1[ka + 1];
            // B VGPR r: K = 2r (r<4) or 8+2r (r>=4), plus half offset
            const int kb = k0 + ((r < 4) ? (2 * r) : (8 + 2 * r)) + bkoff;
            bv[2 * r]     = (__bf16)Bm[(size_t)kb * ldb + ncol];
            bv[2 * r + 1] = (__bf16)Bm[(size_t)(kb + 1) * ldb + ncol];
        }
        acc0 = __builtin_amdgcn_wmma_f32_16x16x32_bf16(
            false, a0, false, bv, (short)0, acc0, false, false);
        acc1 = __builtin_amdgcn_wmma_f32_16x16x32_bf16(
            false, a1, false, bv, (short)0, acc1, false, false);
    }

    // C layout: VGPR r -> row r (+8 for upper lanes), col = ncol
    const int rbase = hi ? 8 : 0;
#pragma unroll
    for (int r = 0; r < 8; ++r) {
        C[(size_t)(rbase + r) * ldc + ncol]      = acc0[r];
        C[(size_t)(16 + rbase + r) * ldc + ncol] = acc1[r];
    }
}

// ---------------------------------------------------------------------------
// RoPE (half-split) applied in place to q and k sections of the qkv buffer.
// grid = B * (NH + NKV), block = 64 (one thread per rotation pair).
// ---------------------------------------------------------------------------
__global__ __launch_bounds__(64) void rope_kernel(float* __restrict__ qkv,
                                                  const int* __restrict__ positions)
{
    const int idx = blockIdx.x;
    const int b   = idx / (NHq + NKVh);
    const int hh  = idx % (NHq + NKVh);
    const int j   = threadIdx.x;       // 0..63

    float* base = qkv + (size_t)b * QKV_COLS +
                  ((hh < NHq) ? hh * Dd : (NHq * Dd + (hh - NHq) * Dd));

    const float pos = (float)positions[b];
    // inv_freq = theta^(-2j/D) = exp(-j * (2/D) * ln(10000))
    const float inv = __expf(-(float)j * (2.0f / (float)Dd) * 9.21034037f);
    const float ang = pos * inv;
    const float c = __cosf(ang);
    const float s = __sinf(ang);
    const float x1 = base[j];
    const float x2 = base[j + 64];
    base[j]      = x1 * c - x2 * s;
    base[j + 64] = x2 * c + x1 * s;
}

// ---------------------------------------------------------------------------
// Flash-decode GQA attention. One workgroup (8 waves) per (batch, kv_head).
// Each wave streams disjoint 32-key tiles with an online softmax and keeps a
// private f32 accumulator (4 query rows x 128 dims as 8 WMMA C tiles).
// Scores: Q(16x32 bf16) x K^T(32x16) WMMAs; P.V: probs(16x32) x V(32x16).
// Appended-token k/v read from the post-RoPE qkv buffer at pos == ctx.
// ---------------------------------------------------------------------------
__global__ __launch_bounds__(256) void attn_kernel(
    const float* __restrict__ qkv,
    const float* __restrict__ kcache,
    const float* __restrict__ vcache,
    const int*   __restrict__ block_tables,
    const int*   __restrict__ context_lens,
    float*       __restrict__ attn_out)
{
    __shared__ float  sQ[16][Dd];           // 8 KB (rows 4..15 zero)
    __shared__ __bf16 sP[8][16][32];        // 8 KB per-wave prob tiles
    __shared__ float  sAcc[8][GRP][Dd];     // 16 KB flash partial sums
    __shared__ float  sM[8][GRP];
    __shared__ float  sL[8][GRP];

    const int b   = blockIdx.x / NKVh;
    const int kvh = blockIdx.x % NKVh;
    const int tid = threadIdx.x;
    const int w   = tid >> 5;
    const int lane = tid & 31;
    const int lo  = lane & 15;
    const int hi  = lane >> 4;

    const int ctx = context_lens[b];
    const int L   = ctx + 1;                // keys 0..ctx inclusive

    // ---- stage Q (post-RoPE) into LDS, zero pad rows 4..15 and sP pad rows
    for (int i = tid; i < 16 * Dd; i += 256) {
        const int g = i >> 7, d = i & 127;
        sQ[g][d] = (g < GRP)
            ? qkv[(size_t)b * QKV_COLS + (kvh * GRP + g) * Dd + d] : 0.0f;
    }
    for (int i = tid; i < 8 * 12 * 32; i += 256) {
        const int ww = i / (12 * 32);
        const int rr = 4 + (i / 32) % 12;
        const int kk = i % 32;
        sP[ww][rr][kk] = (__bf16)0.0f;
    }
    __syncthreads();

    const int*   btab = block_tables + (size_t)b * MBt;
    const float* knew = qkv + (size_t)b * QKV_COLS + NHq * Dd + kvh * Dd;
    const float* vnew = knew + NKVh * Dd;

    // ---- precompute the Q A-operand (constant across key tiles): 4 d-chunks
    v16bf aq[4];
    {
        const int akoff = hi ? 8 : 0;
#pragma unroll
        for (int dc = 0; dc < 4; ++dc) {
#pragma unroll
            for (int r = 0; r < 8; ++r) {
                const int ka = dc * 32 + ((r < 4) ? (2 * r) : (8 + 2 * r)) + akoff;
                aq[dc][2 * r]     = (__bf16)sQ[lo][ka];
                aq[dc][2 * r + 1] = (__bf16)sQ[lo][ka + 1];
            }
        }
    }

    float mg[GRP], lg[GRP];
    v8f   acc[8];
#pragma unroll
    for (int g = 0; g < GRP; ++g) { mg[g] = -1e30f; lg[g] = 0.0f; }
#pragma unroll
    for (int t = 0; t < 8; ++t) acc[t] = (v8f){};

    const float scale = 0.0883883476f;      // 1/sqrt(128)

    for (int pos0 = w * 32; pos0 < L; pos0 += 8 * 32) {
        // ===== scores for 32 keys (two 16-wide WMMA column tiles) =========
        float s0[GRP], s1[GRP];
#pragma unroll
        for (int half = 0; half < 2; ++half) {
            const int p = pos0 + half * 16 + lo;   // this lane's key column
            const float* krow = nullptr;
            if (p < ctx) {
                const int blk = btab[p >> 4];
                krow = kcache + (((size_t)blk * BSz + (p & 15)) * NKVh + kvh) * Dd;
            } else if (p == ctx) {
                krow = knew;
            }
            const int bkoff = hi ? 8 : 0;
            v8f c = {};
#pragma unroll
            for (int dc = 0; dc < 4; ++dc) {
                v16bf bv;
#pragma unroll
                for (int r = 0; r < 8; ++r) {
                    const int d = dc * 32 + bkoff + ((r < 4) ? (2 * r) : (8 + 2 * r));
                    const float f0 = krow ? krow[d]     : 0.0f;
                    const float f1 = krow ? krow[d + 1] : 0.0f;
                    bv[2 * r]     = (__bf16)f0;
                    bv[2 * r + 1] = (__bf16)f1;
                }
                c = __builtin_amdgcn_wmma_f32_16x16x32_bf16(
                    false, aq[dc], false, bv, (short)0, c, false, false);
            }
            // lanes 0-15 hold score[q=r][key=p] in c[r], r=0..3
            const bool valid = (p < L) && (hi == 0);
#pragma unroll
            for (int g = 0; g < GRP; ++g) {
                const float s = valid ? c[g] * scale : -1e30f;
                if (half == 0) s0[g] = s; else s1[g] = s;
            }
        }

        // ===== online softmax (wave32 reductions) ==========================
        float rr[GRP];
#pragma unroll
        for (int g = 0; g < GRP; ++g) {
            float t = fmaxf(s0[g], s1[g]);
#pragma unroll
            for (int off = 1; off < 32; off <<= 1)
                t = fmaxf(t, __shfl_xor(t, off, 32));
            const float mn = fmaxf(mg[g], t);
            rr[g] = __expf(mg[g] - mn);
            mg[g] = mn;
        }
        float p0[GRP], p1[GRP];
#pragma unroll
        for (int g = 0; g < GRP; ++g) {
            p0[g] = __expf(s0[g] - mg[g]);
            p1[g] = __expf(s1[g] - mg[g]);
            float t = p0[g] + p1[g];
#pragma unroll
            for (int off = 1; off < 32; off <<= 1)
                t += __shfl_xor(t, off, 32);
            lg[g] = lg[g] * rr[g] + t;
        }
#pragma unroll
        for (int t = 0; t < 8; ++t)
#pragma unroll
            for (int g = 0; g < GRP; ++g)
                acc[t][g] *= rr[g];

        // ===== re-layout probs C-tile -> A-tile via per-wave LDS bounce ====
        if (hi == 0) {
#pragma unroll
            for (int g = 0; g < GRP; ++g) {
                sP[w][g][lo]      = (__bf16)p0[g];
                sP[w][g][16 + lo] = (__bf16)p1[g];
            }
        }
        // same-wave LDS ops are in-order: read-after-write is safe
        v16bf ap;
        {
            const int akoff = hi ? 8 : 0;
#pragma unroll
            for (int r = 0; r < 8; ++r) {
                const int ka = ((r < 4) ? (2 * r) : (8 + 2 * r)) + akoff;
                ap[2 * r]     = sP[w][lo][ka];
                ap[2 * r + 1] = sP[w][lo][ka + 1];
            }
        }

        // ===== P . V : this lane supplies V rows k = pos0 + {0..7, 16..23} (+8 hi)
        int voff[16];
#pragma unroll
        for (int kk = 0; kk < 16; ++kk) {
            const int k = pos0 + ((kk < 8) ? kk : (8 + kk)) + (hi ? 8 : 0);
            int off = -1;
            if (k < ctx) {
                const int blk = btab[k >> 4];
                off = (int)((((size_t)blk * BSz + (k & 15)) * NKVh + kvh) * Dd);
            }
            voff[kk] = off;          // -1 => past-end (prob already 0) or new token
            if (k == ctx) voff[kk] = -2;
        }
#pragma unroll
        for (int dt = 0; dt < 8; ++dt) {
            const int d = dt * 16 + lo;   // this lane's output column
            v16bf bv;
#pragma unroll
            for (int r = 0; r < 8; ++r) {
                const int o0 = voff[2 * r], o1 = voff[2 * r + 1];
                const float f0 = (o0 >= 0) ? vcache[(size_t)o0 + d]
                                           : ((o0 == -2) ? vnew[d] : 0.0f);
                const float f1 = (o1 >= 0) ? vcache[(size_t)o1 + d]
                                           : ((o1 == -2) ? vnew[d] : 0.0f);
                bv[2 * r]     = (__bf16)f0;
                bv[2 * r + 1] = (__bf16)f1;
            }
            acc[dt] = __builtin_amdgcn_wmma_f32_16x16x32_bf16(
                false, ap, false, bv, (short)0, acc[dt], false, false);
        }
    }

    // ---- publish per-wave flash partials
    if (hi == 0) {
#pragma unroll
        for (int dt = 0; dt < 8; ++dt)
#pragma unroll
            for (int g = 0; g < GRP; ++g)
                sAcc[w][g][dt * 16 + lo] = acc[dt][g];
        if (lo == 0)
#pragma unroll
            for (int g = 0; g < GRP; ++g) { sM[w][g] = mg[g]; sL[w][g] = lg[g]; }
    }
    __syncthreads();

    // ---- merge the 8 waves' partials and write attn output
    for (int i = tid; i < GRP * Dd; i += 256) {
        const int g = i >> 7, d = i & 127;
        float M = -1e30f;
#pragma unroll
        for (int ww = 0; ww < 8; ++ww) M = fmaxf(M, sM[ww][g]);
        float Ls = 0.0f, out = 0.0f;
#pragma unroll
        for (int ww = 0; ww < 8; ++ww) {
            const float f = __expf(sM[ww][g] - M);
            Ls  += sL[ww][g] * f;
            out += sAcc[ww][g][d] * f;
        }
        attn_out[(size_t)b * (NHq * Dd) + (kvh * GRP + g) * Dd + d] = out / Ls;
    }
}

// ---------------------------------------------------------------------------
extern "C" void kernel_launch(void* const* d_in, const int* in_sizes, int n_in,
                              void* d_out, int out_size, void* d_ws, size_t ws_size,
                              hipStream_t stream)
{
    const float* hidden    = (const float*)d_in[0];
    const float* Wqkv      = (const float*)d_in[1];
    const float* Wo        = (const float*)d_in[2];
    const float* kcache    = (const float*)d_in[3];
    const float* vcache    = (const float*)d_in[4];
    const int*   positions = (const int*)d_in[5];
    const int*   btab      = (const int*)d_in[6];
    const int*   ctx       = (const int*)d_in[7];

    float* qkv  = (float*)d_ws;                       // 32*6144 f32
    float* attn = qkv + (size_t)Bb * QKV_COLS;        // 32*4096 f32

    // 1) qkv = hidden @ W_qkv   (M=32, N=6144, K=4096); weights read once
    {
        const int tilesN = QKV_COLS / 16;                  // 384 waves
        wmma_gemm_bf16_m32<<<(tilesN + 3) / 4, 128, 0, stream>>>(
            hidden, Wqkv, qkv, QKV_COLS, Hh, Hh, QKV_COLS, QKV_COLS);
    }
    // 2) RoPE on q and k sections (positions == context_lens)
    rope_kernel<<<Bb * (NHq + NKVh), 64, 0, stream>>>(qkv, positions);

    // 3) paged flash-decode attention
    attn_kernel<<<Bb * NKVh, 256, 0, stream>>>(qkv, kcache, vcache, btab, ctx, attn);

    // 4) out = attn @ W_o   (M=32, N=4096, K=4096); weights read once
    {
        const int tilesN = Hh / 16;                        // 256 waves
        wmma_gemm_bf16_m32<<<(tilesN + 3) / 4, 128, 0, stream>>>(
            attn, Wo, (float*)d_out, Hh, NHq * Dd, NHq * Dd, Hh, Hh);
    }
}
